// MultiHeadAttention_3796751090377
// MI455X (gfx1250) — matrix operand order repaired and checked by reading, not compiled, not run on hardware
//
#include <hip/hip_runtime.h>

#define S_LEN   2048
#define DMODEL  1024
#define NHEADS  16
#define HD      64
#define BATCH   2
#define MROWS   (BATCH * S_LEN)   // 4096
#define SWORDS  (S_LEN / 32)      // 64 packed-mask words per row

typedef __bf16 bf16_t;
typedef bf16_t v16bf __attribute__((ext_vector_type(16)));
typedef bf16_t v8bf  __attribute__((ext_vector_type(8)));
typedef float  v8f   __attribute__((ext_vector_type(8)));

#define SHUF16(lo, hi) __builtin_shufflevector((lo), (hi), 0,1,2,3,4,5,6,7,8,9,10,11,12,13,14,15)
#define WMMA_BF16(a, b, c) \
    __builtin_amdgcn_wmma_f32_16x16x32_bf16(false, (a), false, (b), (short)0, (c), false, false)

__device__ __forceinline__ unsigned short f32_to_bf16(float f) {
    unsigned u = __float_as_uint(f);
    u += 0x7FFFu + ((u >> 16) & 1u);   // round-to-nearest-even
    return (unsigned short)(u >> 16);
}

// ---------------------------------------------------------------------------
// fp32 -> bf16 conversion (vectorized x4)
// ---------------------------------------------------------------------------
__global__ void cvt_kernel(const float* __restrict__ in, unsigned short* __restrict__ out, int n4) {
    int i = blockIdx.x * blockDim.x + threadIdx.x;
    if (i < n4) {
        float4 v = ((const float4*)in)[i];
        uint2 r;
        r.x = (unsigned)f32_to_bf16(v.x) | ((unsigned)f32_to_bf16(v.y) << 16);
        r.y = (unsigned)f32_to_bf16(v.z) | ((unsigned)f32_to_bf16(v.w) << 16);
        ((uint2*)out)[i] = r;
    }
}

// ---------------------------------------------------------------------------
// Pack int32 {0,1} mask into bitmask words: bit t of word w = mask[w*32+t].
// 33 MB -> 1 MB so the attention inner loop reads 1 word per row per 32 keys.
// ---------------------------------------------------------------------------
__global__ void pack_mask_kernel(const int* __restrict__ mask, unsigned* __restrict__ pm, int nwords) {
    int w = blockIdx.x * blockDim.x + threadIdx.x;
    if (w < nwords) {
        const int* src = mask + (size_t)w * 32;
        unsigned word = 0;
#pragma unroll
        for (int t = 0; t < 32; t += 4) {
            int4 v = *(const int4*)(src + t);
            word |= ((unsigned)v.x & 1u) << t;
            word |= ((unsigned)v.y & 1u) << (t + 1);
            word |= ((unsigned)v.z & 1u) << (t + 2);
            word |= ((unsigned)v.w & 1u) << (t + 3);
        }
        pm[w] = word;
    }
}

// ---------------------------------------------------------------------------
// Generic WMMA GEMM: C[m,n] = sum_k A[m,k] * W[n,k]
// A: (M x K) bf16 row-major.  W: (N x K) bf16 row-major (torch Linear weight).
// One wave computes a 64(M) x 32(N) tile: 4 A-frags x 2 B-frags -> 8 WMMA/step.
// MODE 0: bf16 row-major out (Q, K)
// MODE 1: bf16 per-head transposed out: Vt[(b*H+h)*HD + d][s]
// MODE 2: f32 + bias out (final projection)
// ---------------------------------------------------------------------------
template <int MODE>
__global__ void __launch_bounds__(256) gemm_kernel(
    const unsigned short* __restrict__ A,
    const unsigned short* __restrict__ W,
    void* __restrict__ outp,
    const float* __restrict__ bias,
    int M, int N, int K)
{
    const int lane = threadIdx.x & 31;
    const int wave = threadIdx.x >> 5;
    const int tile = blockIdx.x * 8 + wave;
    const int ntn  = N >> 5;            // 32-wide N tiles
    const int tn   = tile % ntn;
    const int tm   = tile / ntn;        // 64-high M tiles
    if (tm * 64 >= M) return;

    const int col = lane & 15;
    const int hg  = lane >> 4;          // 0: lanes 0-15, 1: lanes 16-31
    const int ab  = hg * 16;            // A-fragment K-base

    const unsigned short* arow[4];
#pragma unroll
    for (int ms = 0; ms < 4; ms++)
        arow[ms] = A + (size_t)(tm * 64 + ms * 16 + col) * K;
    const unsigned short* wrow0 = W + (size_t)(tn * 32 + col) * K + hg * 16;
    const unsigned short* wrow1 = wrow0 + (size_t)16 * K;

    v8f c[4][2] = {};

    for (int k0 = 0; k0 < K; k0 += 32) {
        __builtin_prefetch(wrow0 + k0 + 256, 0, 1);
        v16bf b0 = *(const v16bf*)(wrow0 + k0);
        v16bf b1 = *(const v16bf*)(wrow1 + k0);
#pragma unroll
        for (int ms = 0; ms < 4; ms++) {
            __builtin_prefetch(arow[ms] + k0 + 256, 0, 1);
            v8bf lo = *(const v8bf*)(arow[ms] + k0 + ab);
            v8bf hi = *(const v8bf*)(arow[ms] + k0 + ab + 8);
            v16bf a = SHUF16(lo, hi);
            c[ms][0] = WMMA_BF16(a, b0, c[ms][0]);
            c[ms][1] = WMMA_BF16(a, b1, c[ms][1]);
        }
    }

#pragma unroll
    for (int ms = 0; ms < 4; ms++) {
#pragma unroll
        for (int ns = 0; ns < 2; ns++) {
#pragma unroll
            for (int r = 0; r < 8; r++) {
                int m = tm * 64 + ms * 16 + r + 8 * hg;
                int n = tn * 32 + ns * 16 + col;
                float v = c[ms][ns][r];
                if (MODE == 0) {
                    ((unsigned short*)outp)[(size_t)m * N + n] = f32_to_bf16(v);
                } else if (MODE == 1) {
                    int bb = m / S_LEN, s = m % S_LEN;
                    int hh = n >> 6, d = n & 63;
                    ((unsigned short*)outp)[((size_t)(bb * NHEADS + hh) * HD + d) * S_LEN + s] =
                        f32_to_bf16(v);
                } else {
                    ((float*)outp)[(size_t)m * N + n] = v + bias[n];
                }
            }
        }
    }
}

// ---------------------------------------------------------------------------
// Flash-style attention. One wave handles 32 queries for one (b,h); each K/V
// fragment load feeds two WMMAs. Mask read as 1 packed word per row per block.
// ---------------------------------------------------------------------------
__global__ void __launch_bounds__(128) attn_kernel(
    const unsigned short* __restrict__ Q,
    const unsigned short* __restrict__ Kmat,
    const unsigned short* __restrict__ Vt,
    const unsigned* __restrict__ pmask,
    unsigned short* __restrict__ out)
{
    __shared__ __align__(16) unsigned short pstage[4][32 * 32];

    const int lane  = threadIdx.x & 31;
    const int wave  = threadIdx.x >> 5;
    const int qt    = blockIdx.x * 4 + wave;   // 0..63 (32-query tiles)
    const int bh    = blockIdx.y;              // 0..31
    const int b     = bh / NHEADS;
    const int h     = bh % NHEADS;
    const int qbase = qt * 32;
    const int col   = lane & 15;
    const int hg    = lane >> 4;
    const int ab    = hg * 16;

    // Q A-fragments: 2 query sub-tiles x 2 K(d)-steps
    v16bf qa[2][2];
#pragma unroll
    for (int ms = 0; ms < 2; ms++) {
        const unsigned short* qrow =
            Q + (size_t)(b * S_LEN + qbase + ms * 16 + col) * DMODEL + h * HD;
#pragma unroll
        for (int dk = 0; dk < 2; dk++) {
            v8bf lo = *(const v8bf*)(qrow + dk * 32 + ab);
            v8bf hi = *(const v8bf*)(qrow + dk * 32 + ab + 8);
            qa[ms][dk] = SHUF16(lo, hi);
        }
    }

    v8f acc[2][4] = {};                 // 32 x 64 output accumulators
    float mrow[2][8], lrow[2][8];
#pragma unroll
    for (int ms = 0; ms < 2; ms++)
#pragma unroll
        for (int r = 0; r < 8; r++) { mrow[ms][r] = -1e30f; lrow[ms][r] = 0.0f; }

    const unsigned* pmb = pmask + (size_t)(b * S_LEN) * SWORDS;
    unsigned short* ps = pstage[wave];

    for (int j0 = 0; j0 < S_LEN; j0 += 32) {
        const int jw = j0 >> 5;

        // K B-fragments for 32 keys (shared by both query sub-tiles)
        v16bf kb[2][2];
#pragma unroll
        for (int nt = 0; nt < 2; nt++) {
            const unsigned short* krow =
                Kmat + (size_t)(b * S_LEN + j0 + nt * 16 + col) * DMODEL + h * HD + hg * 16;
            kb[nt][0] = *(const v16bf*)(krow);
            kb[nt][1] = *(const v16bf*)(krow + 32);
        }

        float alpha[2][8];
#pragma unroll
        for (int ms = 0; ms < 2; ms++) {
            // scores for this 16-query sub-tile: 16 x 32
            v8f z = {};
            v8f s0 = WMMA_BF16(qa[ms][0], kb[0][0], z);
            s0 = WMMA_BF16(qa[ms][1], kb[0][1], s0);
            v8f s1 = WMMA_BF16(qa[ms][0], kb[1][0], z);
            s1 = WMMA_BF16(qa[ms][1], kb[1][1], s1);

            // scale + bitmask + per-row block max
            float bm[8];
#pragma unroll
            for (int r = 0; r < 8; r++) {
                int i = qbase + ms * 16 + r + 8 * hg;
                unsigned word = pmb[(size_t)i * SWORDS + jw];
                float v0 = s0[r] * 0.125f;     // 1/sqrt(64)
                float v1 = s1[r] * 0.125f;
                if ((word >> col) & 1u)        v0 = -1e9f;
                if ((word >> (16 + col)) & 1u) v1 = -1e9f;
                s0[r] = v0; s1[r] = v1;
                bm[r] = fmaxf(v0, v1);
            }
#pragma unroll
            for (int r = 0; r < 8; r++) {
                float v = bm[r];
                v = fmaxf(v, __shfl_xor(v, 1));
                v = fmaxf(v, __shfl_xor(v, 2));
                v = fmaxf(v, __shfl_xor(v, 4));
                v = fmaxf(v, __shfl_xor(v, 8));
                bm[r] = v;
            }

            // online softmax update
            float rs[8];
#pragma unroll
            for (int r = 0; r < 8; r++) {
                float nm = fmaxf(mrow[ms][r], bm[r]);
                alpha[ms][r] = __expf(mrow[ms][r] - nm);
                mrow[ms][r] = nm;
                float p0 = __expf(s0[r] - nm);
                float p1 = __expf(s1[r] - nm);
                s0[r] = p0; s1[r] = p1;
                rs[r] = p0 + p1;
            }
#pragma unroll
            for (int r = 0; r < 8; r++) {
                float v = rs[r];
                v += __shfl_xor(v, 1);
                v += __shfl_xor(v, 2);
                v += __shfl_xor(v, 4);
                v += __shfl_xor(v, 8);
                lrow[ms][r] = lrow[ms][r] * alpha[ms][r] + v;
            }

            // stage P sub-tile (16x32 bf16) to LDS
#pragma unroll
            for (int r = 0; r < 8; r++) {
                int il = ms * 16 + r + 8 * hg;
                ps[il * 32 + col]      = f32_to_bf16(s0[r]);
                ps[il * 32 + 16 + col] = f32_to_bf16(s1[r]);
            }
        }

        // reload P as A-fragments
        v16bf pa[2];
#pragma unroll
        for (int ms = 0; ms < 2; ms++) {
            const unsigned short* prow = ps + (ms * 16 + col) * 32;
            v8bf lo = *(const v8bf*)(prow + ab);
            v8bf hi = *(const v8bf*)(prow + ab + 8);
            pa[ms] = SHUF16(lo, hi);
        }

        // rescale accumulators, then P·V (each V fragment feeds 2 WMMAs)
#pragma unroll
        for (int ms = 0; ms < 2; ms++)
#pragma unroll
            for (int dt = 0; dt < 4; dt++)
#pragma unroll
                for (int r = 0; r < 8; r++) acc[ms][dt][r] *= alpha[ms][r];
#pragma unroll
        for (int dt = 0; dt < 4; dt++) {
            const unsigned short* vrow =
                Vt + (size_t)(bh * HD + dt * 16 + col) * S_LEN + j0 + hg * 16;
            v16bf vb = *(const v16bf*)(vrow);
            acc[0][dt] = WMMA_BF16(pa[0], vb, acc[0][dt]);
            acc[1][dt] = WMMA_BF16(pa[1], vb, acc[1][dt]);
        }
    }

    // epilogue: normalize, store bf16 (B*S, D) with head interleave
#pragma unroll
    for (int ms = 0; ms < 2; ms++)
#pragma unroll
        for (int r = 0; r < 8; r++) lrow[ms][r] = 1.0f / lrow[ms][r];
#pragma unroll
    for (int ms = 0; ms < 2; ms++) {
#pragma unroll
        for (int dt = 0; dt < 4; dt++) {
#pragma unroll
            for (int r = 0; r < 8; r++) {
                int i = qbase + ms * 16 + r + 8 * hg;
                float o = acc[ms][dt][r] * lrow[ms][r];
                out[(size_t)(b * S_LEN + i) * DMODEL + h * HD + dt * 16 + col] = f32_to_bf16(o);
            }
        }
    }
}

// ---------------------------------------------------------------------------
extern "C" void kernel_launch(void* const* d_in, const int* in_sizes, int n_in,
                              void* d_out, int out_size, void* d_ws, size_t ws_size,
                              hipStream_t stream) {
    const float* x   = (const float*)d_in[0];
    const int*   msk = (const int*)d_in[1];
    const float* Wq  = (const float*)d_in[2];
    const float* Wk  = (const float*)d_in[3];
    const float* Wv  = (const float*)d_in[4];
    const float* Wo  = (const float*)d_in[5];
    const float* bo  = (const float*)d_in[6];
    float* out = (float*)d_out;

    char* ws = (char*)d_ws;
    const size_t MB = 1024u * 1024u;
    unsigned short* xb    = (unsigned short*)(ws + 0);        // 8 MB
    unsigned short* attnb = xb;                               // reused after QKV
    unsigned short* wqb   = (unsigned short*)(ws + 8 * MB);   // 2 MB each
    unsigned short* wkb   = (unsigned short*)(ws + 10 * MB);
    unsigned short* wvb   = (unsigned short*)(ws + 12 * MB);
    unsigned short* wob   = (unsigned short*)(ws + 14 * MB);
    unsigned short* Qb    = (unsigned short*)(ws + 16 * MB);  // 8 MB
    unsigned short* Kb    = (unsigned short*)(ws + 24 * MB);  // 8 MB
    unsigned short* Vtb   = (unsigned short*)(ws + 32 * MB);  // 8 MB
    unsigned*       pmb   = (unsigned*)(ws + 40 * MB);        // 1 MB (peak 41 MB)

    const int nX = MROWS * DMODEL;    // 4 Mi elements
    const int nW = DMODEL * DMODEL;   // 1 Mi elements
    cvt_kernel<<<(nX / 4) / 256, 256, 0, stream>>>(x,  xb,  nX / 4);
    cvt_kernel<<<(nW / 4) / 256, 256, 0, stream>>>(Wq, wqb, nW / 4);
    cvt_kernel<<<(nW / 4) / 256, 256, 0, stream>>>(Wk, wkb, nW / 4);
    cvt_kernel<<<(nW / 4) / 256, 256, 0, stream>>>(Wv, wvb, nW / 4);
    cvt_kernel<<<(nW / 4) / 256, 256, 0, stream>>>(Wo, wob, nW / 4);

    const int nwords = BATCH * S_LEN * SWORDS;                 // 262144
    pack_mask_kernel<<<nwords / 256, 256, 0, stream>>>(msk, pmb, nwords);

    const int gblocks = (MROWS / 64) * (DMODEL / 32) / 8;      // 256
    gemm_kernel<0><<<gblocks, 256, 0, stream>>>(xb, wqb, (void*)Qb,  nullptr, MROWS, DMODEL, DMODEL);
    gemm_kernel<0><<<gblocks, 256, 0, stream>>>(xb, wkb, (void*)Kb,  nullptr, MROWS, DMODEL, DMODEL);
    gemm_kernel<1><<<gblocks, 256, 0, stream>>>(xb, wvb, (void*)Vtb, nullptr, MROWS, DMODEL, DMODEL);

    attn_kernel<<<dim3(S_LEN / 32 / 4, BATCH * NHEADS), 128, 0, stream>>>(Qb, Kb, Vtb, pmb, attnb);

    gemm_kernel<2><<<gblocks, 256, 0, stream>>>(attnb, wob, (void*)out, bo, MROWS, DMODEL, DMODEL);

    (void)in_sizes; (void)n_in; (void)out_size; (void)ws_size;
}
